// SAGEConv_49452253446206
// MI455X (gfx1250) — matrix operand order repaired and checked by reading, not compiled, not run on hardware
//
#include <hip/hip_runtime.h>
#include <hip/hip_bf16.h>

// ---------------------------------------------------------------------------
// GraphSAGE conv for MI455X (gfx1250, wave32, WMMA bf16, async LDS staging).
//
// Algebraic restructure (segment-sum commutes with the linear map W2):
//   Xb    = bf16(X)
//   projb = bf16(Xb @ W3^T + b3)
//   proj2 = bf16(projb @ W2^T)           <- per-node message already projected
//   out   = Xb @ W1^T                    (f32)
//   out[dst] += proj2[src]  for each edge (f32 atomics; operands L2-resident)
//
// Workspace: [Xb bf16][projb bf16][proj2b bf16][w1b|w2b|w3b bf16]  (~77 MB)
// ---------------------------------------------------------------------------

typedef __attribute__((ext_vector_type(16))) __bf16 v16bf;
typedef __attribute__((ext_vector_type(8)))  __bf16 v8bf;
typedef __attribute__((ext_vector_type(8)))  float  v8f;
typedef __attribute__((ext_vector_type(4)))  int    v4i;

#define D 256
constexpr int BSTRIDE = 264;   // 264 bf16 = 528 B row stride -> conflict-free ds_load_b128

#if defined(__HIP_DEVICE_COMPILE__) && __has_builtin(__builtin_amdgcn_global_load_async_to_lds_b128)
#define USE_ASYNC_LDS 1
typedef __attribute__((address_space(1))) v4i* gas_v4i_ptr;   // global int4*
typedef __attribute__((address_space(3))) v4i* las_v4i_ptr;   // LDS int4*
#else
#define USE_ASYNC_LDS 0
#endif

// A-fragment (16x32 bf16): lane m holds row mbase+m; elems 0-7 = K kb..kb+7,
// elems 8-15 = K kb+16..kb+23, kb = k0 + half*8  (two b128 loads, gap 32B).
__device__ __forceinline__ v16bf ld_afrag(const __bf16* __restrict__ p) {
  union { v16bf v; v8bf h[2]; } u;
  u.h[0] = *(const v8bf*)p;
  u.h[1] = *(const v8bf*)(p + 16);
  return u.v;
}

// B-fragment (32x16 bf16): lane m holds col; elems 0-7 = K kb..kb+7,
// elems 8-15 = K kb+16..kb+23, kb = k0 + half*8  (two ds_load_b128 from the staged LDS slab).
__device__ __forceinline__ v16bf ld_bfrag(const __bf16* p) {
  union { v16bf v; v8bf h[2]; } u;
  u.h[0] = *(const v8bf*)p;
  u.h[1] = *(const v8bf*)(p + 16);
  return u.v;
}

#define WMMA_BF16(A_, B_, C_) \
  __builtin_amdgcn_wmma_f32_16x16x32_bf16(false, (A_), false, (B_), (short)0, (C_), false, false)

// ---------------------------------------------------------------------------
// out[M,256] = A[M,256](bf16) @ W[256,256]^T (+bias). Block: 8 waves, each a
// 16x64 strip; B slab (64 cols x 256 K = 32KB) staged in LDS once per block.
// Inner loop is software-pipelined: 4 independent B fragments + next-iter A
// are in flight before the 4 WMMAs issue.
// ---------------------------------------------------------------------------
template <bool HAS_BIAS, bool BF16_OUT>
__global__ __launch_bounds__(256) void k_gemm(
    const __bf16* __restrict__ A, const __bf16* __restrict__ W,
    const float* __restrict__ bias, void* __restrict__ outp, int n_rows) {
  __shared__ __bf16 Bs[64 * BSTRIDE];
  const int nbase = blockIdx.x * 64;

  // ---- stage W[nbase..nbase+64) x K256 into LDS (async path on gfx1250) ----
  {
    const int t = threadIdx.x;
#pragma unroll
    for (int i = 0; i < 8; ++i) {
      const int c   = t + i * 256;      // 2048 chunks of 8 bf16 (16B)
      const int row = c >> 5;           // 32 chunks per 512B row
      const int k8  = (c & 31) * 8;
      const __bf16* g = W + (size_t)(nbase + row) * D + k8;
      __bf16*       l = &Bs[row * BSTRIDE + k8];
#if USE_ASYNC_LDS
      __builtin_amdgcn_global_load_async_to_lds_b128(
          (gas_v4i_ptr)(uintptr_t)g,
          (las_v4i_ptr)(unsigned)(uintptr_t)l, 0, 0);
#else
      *(v8bf*)l = *(const v8bf*)g;
#endif
    }
#if USE_ASYNC_LDS
#if __has_builtin(__builtin_amdgcn_s_wait_asynccnt)
    __builtin_amdgcn_s_wait_asynccnt(0);
#else
    asm volatile("s_wait_asynccnt 0x0" ::: "memory");
#endif
#endif
    __syncthreads();
  }

  const int lane  = threadIdx.x & 31;
  const int wave  = threadIdx.x >> 5;
  const int half  = lane >> 4;
  const int m     = lane & 15;
  const int mbase = blockIdx.y * 128 + wave * 16;
  const int arow  = mbase + m;
  const int arowc = arow < n_rows ? arow : (n_rows - 1);   // clamp: EXEC must stay all-1s
  const __bf16* Ap = A + (size_t)arowc * D + half * 8;

  v8f c0, c1, c2, c3;
  if constexpr (HAS_BIAS) {
    const float bi0 = bias[nbase + m],      bi1 = bias[nbase + 16 + m];
    const float bi2 = bias[nbase + 32 + m], bi3 = bias[nbase + 48 + m];
#pragma unroll
    for (int r = 0; r < 8; ++r) { c0[r] = bi0; c1[r] = bi1; c2[r] = bi2; c3[r] = bi3; }
  } else {
    c0 = v8f{}; c1 = v8f{}; c2 = v8f{}; c3 = v8f{};
  }

  // ---- software-pipelined K loop ----
  v16bf a = ld_afrag(Ap);                     // prologue: first A fragment
#pragma unroll
  for (int k0 = 0; k0 < D; k0 += 32) {
    const __bf16* bp = &Bs[(size_t)m * BSTRIDE + k0 + half * 8];
    const v16bf b0 = ld_bfrag(bp);
    const v16bf b1 = ld_bfrag(bp + 16 * BSTRIDE);
    const v16bf b2 = ld_bfrag(bp + 32 * BSTRIDE);
    const v16bf b3 = ld_bfrag(bp + 48 * BSTRIDE);
    v16bf an = a;
    if (k0 + 32 < D) an = ld_afrag(Ap + (k0 + 32));   // prefetch next A
    c0 = WMMA_BF16(a, b0, c0);
    c1 = WMMA_BF16(a, b1, c1);
    c2 = WMMA_BF16(a, b2, c2);
    c3 = WMMA_BF16(a, b3, c3);
    a = an;
  }

  // C layout: VGPR r, lane l -> row = mbase + r + half*8, col = nbase + (l&15)
  const int crowb = mbase + half * 8;
#pragma unroll
  for (int r = 0; r < 8; ++r) {
    const int cr = crowb + r;
    if (cr < n_rows) {
      if constexpr (BF16_OUT) {
        __bf16* o = (__bf16*)outp + (size_t)cr * D + nbase + m;
        o[0] = (__bf16)c0[r]; o[16] = (__bf16)c1[r];
        o[32] = (__bf16)c2[r]; o[48] = (__bf16)c3[r];
      } else {
        float* o = (float*)outp + (size_t)cr * D + nbase + m;
        o[0] = c0[r]; o[16] = c1[r]; o[32] = c2[r]; o[48] = c3[r];
      }
    }
  }
}

// ---- fp32 -> bf16 bulk convert (8 elems/thread, b128 in / b128 out) --------
__global__ void k_cvt_f32_bf16(const float* __restrict__ src,
                               __bf16* __restrict__ dst, int n8) {
  const int i = blockIdx.x * blockDim.x + threadIdx.x;
  if (i >= n8) return;
  const float4* s = (const float4*)(src + (size_t)i * 8);
  const float4 f0 = s[0], f1 = s[1];
  v8bf o;
  o[0] = (__bf16)f0.x; o[1] = (__bf16)f0.y; o[2] = (__bf16)f0.z; o[3] = (__bf16)f0.w;
  o[4] = (__bf16)f1.x; o[5] = (__bf16)f1.y; o[6] = (__bf16)f1.z; o[7] = (__bf16)f1.w;
  *(v8bf*)(dst + (size_t)i * 8) = o;
}

// ---- weights fp32 -> bf16 (3 x 256x256 packed) -----------------------------
__global__ void k_cvt_w(const float* __restrict__ w1, const float* __restrict__ w2,
                        const float* __restrict__ w3, __bf16* __restrict__ wb) {
  const int i = blockIdx.x * blockDim.x + threadIdx.x;
  const int NW = D * D;
  float v;
  if (i < NW)          v = w1[i];
  else if (i < 2 * NW) v = w2[i - NW];
  else                 v = w3[i - 2 * NW];
  wb[i] = (__bf16)v;
}

// ---- edge scatter: one wave per edge; per lane one b128 gather (8 bf16),
// 8 global_atomic_add_f32 into out[dst]. proj2b (25.6MB) is L2-resident. ----
__global__ __launch_bounds__(256) void k_scatter(
    const long long* __restrict__ ei, const __bf16* __restrict__ proj2,
    float* __restrict__ out, int n_edges) {
  const int gwave = (int)((blockIdx.x * blockDim.x + threadIdx.x) >> 5);
  const int lane  = threadIdx.x & 31;
  if (gwave >= n_edges) return;
  const long long src = ei[gwave];
  const long long dst = ei[(size_t)n_edges + gwave];
  const v8bf p = *(const v8bf*)(proj2 + (size_t)src * D + lane * 8);
  float*     q = out + (size_t)dst * D + lane * 8;
#pragma unroll
  for (int i = 0; i < 8; ++i) {
    unsafeAtomicAdd(q + i, (float)p[i]);   // global_atomic_add_f32
  }
}

extern "C" void kernel_launch(void* const* d_in, const int* in_sizes, int n_in,
                              void* d_out, int out_size, void* d_ws, size_t ws_size,
                              hipStream_t stream) {
  const float*     X  = (const float*)d_in[0];
  const long long* ei = (const long long*)d_in[1];   // int64 [2, E]
  const float*     w1 = (const float*)d_in[2];
  const float*     w2 = (const float*)d_in[3];
  const float*     w3 = (const float*)d_in[4];
  const float*     b3 = (const float*)d_in[5];

  const int n = in_sizes[0] / D;   // 50000
  const int e = in_sizes[1] / 2;   // 320000

  // workspace carve-up (~77 MB)
  char*   ws     = (char*)d_ws;
  __bf16* Xb     = (__bf16*)ws;
  __bf16* projb  = Xb + (size_t)n * D;
  __bf16* proj2b = projb + (size_t)n * D;
  __bf16* wb     = proj2b + (size_t)n * D;
  __bf16* w1b = wb, *w2b = wb + D * D, *w3b = wb + 2 * D * D;

  // 1) weights + features -> bf16
  k_cvt_w<<<(3 * D * D) / 256, 256, 0, stream>>>(w1, w2, w3, wb);
  const int n8 = (n * D) / 8;
  k_cvt_f32_bf16<<<(n8 + 255) / 256, 256, 0, stream>>>(X, Xb, n8);

  dim3 grid(D / 64, (n + 127) / 128);   // (4, 391)

  // 2) projb = bf16(Xb @ W3^T + b3)
  k_gemm<true, true><<<grid, 256, 0, stream>>>(Xb, w3b, b3, projb, n);

  // 3) proj2b = bf16(projb @ W2^T)   (message already through W2)
  k_gemm<false, true><<<grid, 256, 0, stream>>>(projb, w2b, nullptr, proj2b, n);

  // 4) out = Xb @ W1^T   (f32)
  k_gemm<false, false><<<grid, 256, 0, stream>>>(Xb, w1b, nullptr, d_out, n);

  // 5) out[dst] += proj2b[src] over edges
  k_scatter<<<(e + 7) / 8, 256, 0, stream>>>(ei, proj2b, (float*)d_out, e);
}
